// ModalityConsisLoss_11132555231828
// MI455X (gfx1250) — matrix operand order repaired and checked by reading, not compiled, not run on hardware
//
#include <hip/hip_runtime.h>
#include <hip/hip_bf16.h>

typedef __attribute__((ext_vector_type(16))) __bf16 v16bf;
typedef __attribute__((ext_vector_type(8)))  __bf16 v8bf;
typedef __attribute__((ext_vector_type(8)))  float  v8f;
typedef __attribute__((ext_vector_type(4)))  unsigned int v4u;
typedef __attribute__((ext_vector_type(4)))  int v4i;
typedef __attribute__((ext_vector_type(8)))  int v8i;

#define B_      2048
#define D_      512
#define TWO_D_  1024
#define HALF_   6144      /* 3*B   */
#define ROWS_   12288     /* 6*B   */
#define TEMP_   0.5f
#define EXP2C_  2.8853900817779268f   /* log2(e)/TEMP */
#define LN2_    0.6931471805599453f
#define NCHUNK  (ROWS_ / 32)
#define BSROW_  528                    /* 512 + 16 pad elements -> 1056B rows */

#define SHUF16(lo, hi) __builtin_shufflevector(lo, hi, 0,1,2,3,4,5,6,7,8,9,10,11,12,13,14,15)

// ---- staging-path selection --------------------------------------------------
#if __has_builtin(__builtin_amdgcn_tensor_load_to_lds)
  #define USE_TDM 1
#elif __has_builtin(__builtin_amdgcn_global_load_async_to_lds_b128)
  #define USE_ASYNC 1
#endif

__device__ __forceinline__ void stage_wait(bool has_next) {
#if defined(USE_TDM)
  #if __has_builtin(__builtin_amdgcn_s_wait_tensorcnt)
    if (has_next) __builtin_amdgcn_s_wait_tensorcnt(1);
    else          __builtin_amdgcn_s_wait_tensorcnt(0);
  #else
    if (has_next) asm volatile("s_wait_tensorcnt 0x1" ::: "memory");
    else          asm volatile("s_wait_tensorcnt 0x0" ::: "memory");
  #endif
#elif defined(USE_ASYNC)
  #if __has_builtin(__builtin_amdgcn_s_wait_asynccnt)
    if (has_next) __builtin_amdgcn_s_wait_asynccnt(8);
    else          __builtin_amdgcn_s_wait_asynccnt(0);
  #else
    if (has_next) asm volatile("s_wait_asynccnt 0x8" ::: "memory");
    else          asm volatile("s_wait_asynccnt 0x0" ::: "memory");
  #endif
#else
  (void)has_next;
#endif
}

// Stage one 32-row (32KB) chunk of z into a padded LDS buffer.
// TDM: single tensor_load_to_lds issued by wave 0; HW inserts 8 DWORDs of pad
//      every 256 DWORDs (-> 1056B LDS rows).  ASYNC/sync: per-thread copy.
__device__ __forceinline__ void stage_chunk(const __bf16* __restrict__ gsrc,
                                            __bf16* __restrict__ lbase,
                                            int tid, int wave) {
#if defined(USE_TDM)
    if (wave == 0) {
        const unsigned long long ga = (unsigned long long)(size_t)gsrc;
        const unsigned lds_addr = (unsigned)(size_t)lbase;   // low 32 bits = LDS byte offset
        v4u g0;
        g0.x = 0x1u;                                          // count=1, user mode
        g0.y = lds_addr;                                      // D#.lds_addr
        g0.z = (unsigned)(ga & 0xffffffffu);                  // global_addr[31:0]
        g0.w = (unsigned)((ga >> 32) & 0x01ffffffu)           // global_addr[56:32]
               | 0x80000000u;                                 // type = 2 ("image")
        v8i g1;
        g1[0] = 0x0FD10000;     // data_size=1(2B) | pad_enable | pad_interval=7(256dw) | pad_amount=7(8dw)
        g1[1] = 0x40000000;     // tensor_dim0[15:0]=16384 in bits[63:48]
        g1[2] = 0x00010000;     // tensor_dim0[31:16]=0 | tensor_dim1=1
        g1[3] = 0x40000000;     // tile_dim0 = 16384 elements (32KB)
        g1[4] = 0x00000001;     // tile_dim1 = 1, tile_dim2 = 0
        g1[5] = 16384;          // tensor_dim0_stride[31:0]
        g1[6] = 0;
        g1[7] = 0;
        v4i gz = {0, 0, 0, 0};
  #if defined(__clang_major__) && (__clang_major__ >= 23)
        v8i gz8 = {0, 0, 0, 0, 0, 0, 0, 0};
        __builtin_amdgcn_tensor_load_to_lds(g0, g1, gz, gz, gz8, 0);
  #else
        __builtin_amdgcn_tensor_load_to_lds(g0, g1, gz, gz, 0);
  #endif
    }
#elif defined(USE_ASYNC)
    const int cr   = tid >> 3;                 // row 0..31
    const int cseg = tid & 7;                  // 128B segment of the 1KB row
    const char* gp = (const char*)(gsrc + (size_t)cr * D_) + cseg * 128;
    char*       lp = (char*)(lbase + (size_t)cr * BSROW_) + cseg * 128;
#pragma unroll
    for (int j = 0; j < 8; ++j) {
        __builtin_amdgcn_global_load_async_to_lds_b128(
            (const __attribute__((address_space(1))) void*)(gp + j * 16),
            (__attribute__((address_space(3)))       void*)(lp + j * 16), 0, 0);
    }
#else
    const int cr   = tid >> 3;
    const int cseg = tid & 7;
    const uint4* src = (const uint4*)(gsrc + (size_t)cr * D_);
    uint4*       dst = (uint4*)(lbase + (size_t)cr * BSROW_);
#pragma unroll
    for (int j = 0; j < 8; ++j) dst[cseg * 8 + j] = src[cseg * 8 + j];
#endif
}

// ---------------------------------------------------------------- init ------
__global__ void __launch_bounds__(256) mcl_init_kernel(float* __restrict__ sumsq,
                                                       float* __restrict__ out) {
    int i = blockIdx.x * blockDim.x + threadIdx.x;
    if (i < ROWS_) sumsq[i] = 0.0f;
    if (i == 0)    out[0]   = 0.0f;
}

// ------------------------------------------------------------ projection ----
__global__ void __launch_bounds__(256) mcl_proj_kernel(const float* __restrict__ fspa,
                                                       const float* __restrict__ fseq,
                                                       const float* __restrict__ W,
                                                       const float* __restrict__ bvec,
                                                       __bf16* __restrict__ z,
                                                       float* __restrict__ sumsq) {
    __shared__ __bf16 As[32][40];
    __shared__ __bf16 Bs[64][40];

    const int tid  = threadIdx.x;
    const int lane = tid & 31;
    const int wave = tid >> 5;
    const int ln   = lane & 15;
    const int lh   = lane >> 4;
    const int wr   = wave >> 2;
    const int wc   = wave & 3;

    const int rowbase = blockIdx.y * 32;
    const int colbase = blockIdx.x * 64;

    const int sa_r  = tid >> 3;
    const int sa_k4 = (tid & 7) * 4;
    const int gr    = rowbase + sa_r;
    const float* fbase = (gr < HALF_) ? fspa : fseq;
    const int rr  = (gr < HALF_) ? gr : (gr - HALF_);
    const int bi  = rr & (B_ - 1);
    const int pi  = rr >> 11;
    const int sel = (pi == 2) ? 3 : pi;
    const float* rowL = fbase + (size_t)(bi * 4 + sel) * D_;
    const float* rowR = fbase + (size_t)(bi * 4 + 2)   * D_;

    const int sb_k  = tid >> 3;
    const int sb_n8 = (tid & 7) * 8;

    v8f acc = {};

    for (int kt = 0; kt < 32; ++kt) {
        const int kb = kt * 32;
        const float* asrc = (kb < D_) ? (rowL + kb + sa_k4)
                                      : (rowR + (kb - D_) + sa_k4);
        float4 a4 = *(const float4*)asrc;
        As[sa_r][sa_k4 + 0] = (__bf16)a4.x;
        As[sa_r][sa_k4 + 1] = (__bf16)a4.y;
        As[sa_r][sa_k4 + 2] = (__bf16)a4.z;
        As[sa_r][sa_k4 + 3] = (__bf16)a4.w;
        const float* wsrc = W + (size_t)(kb + sb_k) * D_ + colbase + sb_n8;
        float4 w0 = *(const float4*)(wsrc + 0);
        float4 w1 = *(const float4*)(wsrc + 4);
        Bs[sb_n8 + 0][sb_k] = (__bf16)w0.x;
        Bs[sb_n8 + 1][sb_k] = (__bf16)w0.y;
        Bs[sb_n8 + 2][sb_k] = (__bf16)w0.z;
        Bs[sb_n8 + 3][sb_k] = (__bf16)w0.w;
        Bs[sb_n8 + 4][sb_k] = (__bf16)w1.x;
        Bs[sb_n8 + 5][sb_k] = (__bf16)w1.y;
        Bs[sb_n8 + 6][sb_k] = (__bf16)w1.z;
        Bs[sb_n8 + 7][sb_k] = (__bf16)w1.w;
        __syncthreads();

        const __bf16* arow = &As[wr * 16 + ln][0];
        v8bf a_lo = *(const v8bf*)(arow + lh * 8);
        v8bf a_hi = *(const v8bf*)(arow + 16 + lh * 8);
        v16bf afr = SHUF16(a_lo, a_hi);
        const __bf16* brow = &Bs[wc * 16 + ln][0];
        v8bf b_lo = *(const v8bf*)(brow + lh * 8);
        v8bf b_hi = *(const v8bf*)(brow + 16 + lh * 8);
        v16bf bfr = SHUF16(b_lo, b_hi);

        acc = __builtin_amdgcn_wmma_f32_16x16x32_bf16(
                  false, afr, false, bfr, (short)0, acc, false, false);
        __syncthreads();
    }

    const int Cn = colbase + wc * 16 + ln;
    const float bc = bvec[Cn];
#pragma unroll
    for (int v = 0; v < 8; ++v) {
        const int M = v + lh * 8;
        const int R = rowbase + wr * 16 + M;
        float val = acc[v] + bc;
        z[(size_t)R * D_ + Cn] = (__bf16)val;
        float ss = val * val;
        ss += __shfl_xor(ss, 1, 32);
        ss += __shfl_xor(ss, 2, 32);
        ss += __shfl_xor(ss, 4, 32);
        ss += __shfl_xor(ss, 8, 32);
        if (ln == 0) unsafeAtomicAdd(&sumsq[R], ss);
    }
}

// ------------------------------------------------------------ row norms ----
__global__ void __launch_bounds__(256) mcl_norm_kernel(const float* __restrict__ sumsq,
                                                       float* __restrict__ rnorm) {
    int i = blockIdx.x * blockDim.x + threadIdx.x;
    if (i < ROWS_) {
        float n = __builtin_sqrtf(sumsq[i]);
        rnorm[i] = 1.0f / fmaxf(n, 1e-12f);
    }
}

// ---------------------------------------------------- fused sim + logsumexp -
// 64 rows per block; A-fragments register-resident; 32-col chunks staged into
// double-buffered LDS by the Tensor Data Mover, overlapped with WMMA compute.
__global__ void __launch_bounds__(256, 1) mcl_sim_kernel(const __bf16* __restrict__ z,
                                                         const float* __restrict__ rnorm,
                                                         float* __restrict__ out) {
    __shared__ __bf16 Bs[2][32][BSROW_];   // 2 x 33KB, 1056B rows (32B pad)

    const int tid  = threadIdx.x;
    const int lane = tid & 31;
    const int wave = tid >> 5;
    const int ln   = lane & 15;
    const int lh   = lane >> 4;
    const int wr   = wave >> 1;
    const int wc   = wave & 1;
    const int rowbase = blockIdx.x * 64;

    const int Ra = rowbase + wr * 16 + ln;
    const __bf16* zrow = z + (size_t)Ra * D_;
    v16bf afrag[16];
#pragma unroll
    for (int kt = 0; kt < 16; ++kt) {
        v8bf lo = *(const v8bf*)(zrow + kt * 32 + lh * 8);
        v8bf hi = *(const v8bf*)(zrow + kt * 32 + 16 + lh * 8);
        afrag[kt] = SHUF16(lo, hi);
    }
    float rnA[8];
#pragma unroll
    for (int v = 0; v < 8; ++v) rnA[v] = rnorm[rowbase + wr * 16 + lh * 8 + v];

    float rowsum[8] = {}, diag[8] = {}, possim[8] = {};

    // prologue: stage chunk 0 into buffer 0
    stage_chunk(z, &Bs[0][0][0], tid, wave);

    for (int i = 0; i < NCHUNK; ++i) {
        const int cb  = i * 32;
        const int cur = i & 1;
        const bool has_next = (i + 1) < NCHUNK;
        if (has_next)
            stage_chunk(z + (size_t)(cb + 32) * D_, &Bs[cur ^ 1][0][0], tid, wave);
        const float rnC = rnorm[cb + wc * 16 + ln];
        stage_wait(has_next);            // chunk i resident (TENSORcnt/ASYNCcnt)
        __syncthreads();                 // visible to all waves

        v8f acc = {};
#pragma unroll
        for (int kt = 0; kt < 16; ++kt) {
            const __bf16* brow = &Bs[cur][wc * 16 + ln][kt * 32 + lh * 8];
            v8bf lo = *(const v8bf*)(brow);
            v8bf hi = *(const v8bf*)(brow + 16);
            v16bf bfr = SHUF16(lo, hi);
            acc = __builtin_amdgcn_wmma_f32_16x16x32_bf16(
                      false, afrag[kt], false, bfr, (short)0, acc, false, false);
        }

        const int jn = cb + wc * 16 + ln;
#pragma unroll
        for (int v = 0; v < 8; ++v) {
            const int im  = rowbase + wr * 16 + lh * 8 + v;
            const float sim = acc[v] * rnA[v] * rnC;
            const float e   = __builtin_amdgcn_exp2f(sim * EXP2C_);
            rowsum[v] += e;
            diag[v]   += (jn == im) ? e : 0.0f;
            const int pim = (im < HALF_) ? (im + HALF_) : (im - HALF_);
            possim[v] += (jn == pim) ? sim : 0.0f;
        }
        __syncthreads();                 // guard buffer reuse before next DMA lands
    }

    float lloss = 0.0f;
#pragma unroll
    for (int v = 0; v < 8; ++v) {
        float rs = rowsum[v], dg = diag[v], ps = possim[v];
        rs += __shfl_xor(rs, 1, 32); rs += __shfl_xor(rs, 2, 32);
        rs += __shfl_xor(rs, 4, 32); rs += __shfl_xor(rs, 8, 32);
        dg += __shfl_xor(dg, 1, 32); dg += __shfl_xor(dg, 2, 32);
        dg += __shfl_xor(dg, 4, 32); dg += __shfl_xor(dg, 8, 32);
        ps += __shfl_xor(ps, 1, 32); ps += __shfl_xor(ps, 2, 32);
        ps += __shfl_xor(ps, 4, 32); ps += __shfl_xor(ps, 8, 32);
        if (ln == 0) {
            const float denom = rs - dg;
            const float lg = __builtin_amdgcn_logf(denom) * LN2_;
            lloss += lg - ps * (1.0f / TEMP_);
        }
    }
    if (ln == 0) unsafeAtomicAdd(out, lloss * (1.0f / (float)ROWS_));
}

// ---------------------------------------------------------------- launch ----
extern "C" void kernel_launch(void* const* d_in, const int* in_sizes, int n_in,
                              void* d_out, int out_size, void* d_ws, size_t ws_size,
                              hipStream_t stream) {
    const float* f_seq = (const float*)d_in[0];
    const float* f_spa = (const float*)d_in[1];
    const float* W     = (const float*)d_in[2];
    const float* bvec  = (const float*)d_in[3];
    float* out = (float*)d_out;

    __bf16* z = (__bf16*)d_ws;
    const size_t zbytes = (size_t)ROWS_ * D_ * sizeof(__bf16);
    float* sumsq = (float*)((char*)d_ws + zbytes);
    float* rnorm = sumsq + ROWS_;

    mcl_init_kernel<<<ROWS_ / 256, 256, 0, stream>>>(sumsq, out);

    dim3 gProj(D_ / 64, ROWS_ / 32);
    mcl_proj_kernel<<<gProj, 256, 0, stream>>>(f_spa, f_seq, W, bvec, z, sumsq);

    mcl_norm_kernel<<<ROWS_ / 256, 256, 0, stream>>>(sumsq, rnorm);

    mcl_sim_kernel<<<ROWS_ / 64, 256, 0, stream>>>(z, rnorm, out);
}
